// BasicRecurrentEntityEncoder_62268435857626
// MI455X (gfx1250) — matrix operand order repaired and checked by reading, not compiled, not run on hardware
//
#include <hip/hip_runtime.h>
#include <math.h>

// Problem constants (from reference): VOCAB=50000, D=256, B=128, S=64, L=32, K=32
#define DD   256
#define BB   128
#define SS   64
#define LL   32
#define KK   32
#define NROW (BB * KK)   // 4096 rows of h
#define NSENT (BB * SS)  // 8192 sentences

#define HSTR 260         // padded f32 row stride for h tiles in LDS
#define USTR 272         // padded bf16 row stride for U^T in LDS (544B, 32B aligned)

typedef __attribute__((ext_vector_type(16))) __bf16 bf16x16;
typedef __attribute__((ext_vector_type(8)))  float  floatx8;

// ---------------------------------------------------------------------------
// Kernel 1: enc_sents[b,s,:] = sum_l emb[prgrph[b,s,l], :]
// ---------------------------------------------------------------------------
__global__ void embed_sum_kernel(const int* __restrict__ prgrph,
                                 const float* __restrict__ emb,
                                 float* __restrict__ enc) {
  __shared__ int ids[LL];
  const int bs  = blockIdx.x;      // 0..B*S-1
  const int tid = threadIdx.x;     // 0..255 == d
  if (tid < LL) ids[tid] = prgrph[(size_t)bs * LL + tid];
  __syncthreads();
  float acc = 0.f;
#pragma unroll 4
  for (int l = 0; l < LL; ++l)
    acc += emb[(size_t)ids[l] * DD + tid];
  enc[(size_t)bs * DD + tid] = acc;
}

// ---------------------------------------------------------------------------
// Shared WMMA tile compute: one wave computes a 16x32 slice of (16 rows) @ U.
// sH: 16 x DD f32 (stride HSTR). sUt: U^T as bf16, [n][k], stride USTR.
// Wave `wave` produces columns [wave*32, wave*32+32) in acc0 (n..n+15) and
// acc1 (n+16..n+31). A and B fragments both use contiguous k = 16*half + s (wave32).
// ---------------------------------------------------------------------------
__device__ __forceinline__ void wmma_row_tile(const float* sH_,
                                              const __bf16* sUt_,
                                              int lane, int wave,
                                              floatx8& acc0, floatx8& acc1) {
  const int half = lane >> 4;      // half-wave
  const int m    = lane & 15;      // A-matrix row for this lane
  const int nl   = lane & 15;      // B/C column within tile
  const int n0   = wave * 32;
  floatx8 a0 = {0.f, 0.f, 0.f, 0.f, 0.f, 0.f, 0.f, 0.f};
  floatx8 a1 = {0.f, 0.f, 0.f, 0.f, 0.f, 0.f, 0.f, 0.f};
#pragma unroll
  for (int kk = 0; kk < 8; ++kk) {           // K = 256 in chunks of 32
    // A fragment, 16-bit 16x32: lane holds 16 consecutive K values (by half-wave).
    bf16x16 a;
    const int kb0 = kk * 32 + half * 16;
    const int kb1 = kk * 32 + half * 16 + 8;
#pragma unroll
    for (int i = 0; i < 8; ++i) a[i]     = (__bf16)sH_[m * HSTR + kb0 + i];
#pragma unroll
    for (int i = 0; i < 8; ++i) a[8 + i] = (__bf16)sH_[m * HSTR + kb1 + i];
    // B fragments: lane = column n, 16 consecutive K values (by half-wave).
    const int kb = kk * 32 + half * 16;
    bf16x16 b0 = *(const bf16x16*)&sUt_[(size_t)(n0 + nl) * USTR + kb];
    bf16x16 b1 = *(const bf16x16*)&sUt_[(size_t)(n0 + 16 + nl) * USTR + kb];
    a0 = __builtin_amdgcn_wmma_f32_16x16x32_bf16(false, a, false, b0,
                                                 (short)0, a0, false, false);
    a1 = __builtin_amdgcn_wmma_f32_16x16x32_bf16(false, a, false, b1,
                                                 (short)0, a1, false, false);
  }
  acc0 = a0;
  acc1 = a1;
}

// ---------------------------------------------------------------------------
// Kernel 2: C[M x 256] = A[M x 256] (f32) @ Bm[256 x 256] (f32), bf16 WMMA.
// One block per 16-row tile; 256 threads (8 waves), each wave 32 columns.
// ---------------------------------------------------------------------------
__global__ void __launch_bounds__(256, 1)
gemm256_bf16_kernel(const float* __restrict__ A, const float* __restrict__ Bm,
                    float* __restrict__ C) {
  __shared__ __attribute__((aligned(32))) __bf16 sBt[DD * USTR];
  __shared__ __attribute__((aligned(16))) float  sA[16 * HSTR];
  const int tid  = threadIdx.x;
  const int tile = blockIdx.x;
  // Stage B transposed as bf16: sBt[n][k] = Bm[k][n]
  for (int idx = tid; idx < DD * DD; idx += 256) {
    const int k = idx >> 8, n = idx & 255;
    sBt[(size_t)n * USTR + k] = (__bf16)Bm[idx];
  }
  // Stage 16-row A tile
  for (int idx = tid; idx < 16 * DD; idx += 256) {
    const int r = idx >> 8, c = idx & 255;
    sA[r * HSTR + c] = A[(size_t)(tile * 16 + r) * DD + c];
  }
  __syncthreads();
  const int wave = tid >> 5, lane = tid & 31;
  floatx8 acc0, acc1;
  wmma_row_tile(sA, sBt, lane, wave, acc0, acc1);
  const int half = lane >> 4, nl = lane & 15;
#pragma unroll
  for (int j = 0; j < 8; ++j) {
    const int m = j + 8 * half;
    C[(size_t)(tile * 16 + m) * DD + wave * 32 + nl]      = acc0[j];
    C[(size_t)(tile * 16 + m) * DD + wave * 32 + 16 + nl] = acc1[j];
  }
}

// ---------------------------------------------------------------------------
// Kernel 3: the 64-step recurrence. One block owns 16 rows of h (one WMMA
// M-tile, all within one batch b since K=32). h lives in LDS for all steps;
// U lives in LDS as bf16^T. No global synchronization anywhere.
// ---------------------------------------------------------------------------
__global__ void __launch_bounds__(256, 1)
entity_scan_kernel(const float* __restrict__ keys,   // (B*K, D)
                   const float* __restrict__ keysV,  // (B*K, D)  = keys @ V
                   const float* __restrict__ enc,    // (B*S, D)  = enc_sents
                   const float* __restrict__ eW,     // (B*S, D)  = enc @ W
                   const unsigned char* __restrict__ mask, // (B,S,L) bool
                   const float* __restrict__ U,      // (D, D)
                   float* __restrict__ out)          // (B*K, D)
{
  __shared__ __attribute__((aligned(32))) __bf16 sUt[DD * USTR]; // ~136 KB
  __shared__ __attribute__((aligned(16))) float  sH[16 * HSTR];  // ~16 KB
  __shared__ float sE[DD];
  __shared__ float sEW[DD];
  __shared__ float sGate[16];
  __shared__ float sNorm[16];
  __shared__ int   sMask;

  const int tid  = threadIdx.x;
  const int tile = blockIdx.x;        // 0..255
  const int row0 = tile * 16;         // global h row base
  const int b    = row0 >> 5;         // K = 32 -> tile/2

  // Stage U^T as bf16 (read 64x per step loop -> LDS resident).
  for (int idx = tid; idx < DD * DD; idx += 256) {
    const int k = idx >> 8, n = idx & 255;
    sUt[(size_t)n * USTR + k] = (__bf16)U[idx];
  }
  // h0 = 0
  for (int idx = tid; idx < 16 * HSTR; idx += 256) sH[idx] = 0.f;

  // Per-thread constants for the gate dot: keys[row][c], c = gl + 16*i
  const int grow = tid >> 4;          // 0..15: row within tile
  const int gl   = tid & 15;
  float kreg[16];
#pragma unroll
  for (int i = 0; i < 16; ++i)
    kreg[i] = keys[(size_t)(row0 + grow) * DD + gl + 16 * i];

  // Per-lane constants for the epilogue: keysV slice (fixed across steps).
  const int wave = tid >> 5, lane = tid & 31;
  const int half = lane >> 4, nl = lane & 15;
  float kv0[8], kv1[8];
#pragma unroll
  for (int j = 0; j < 8; ++j) {
    const int m = j + 8 * half;
    kv0[j] = keysV[(size_t)(row0 + m) * DD + wave * 32 + nl];
    kv1[j] = keysV[(size_t)(row0 + m) * DD + wave * 32 + 16 + nl];
  }
  __syncthreads();

  for (int s = 0; s < SS; ++s) {
    // ---- stage this step's e, eW, mask ----
    sE[tid]  = enc[((size_t)b * SS + s) * DD + tid];
    sEW[tid] = eW[((size_t)b * SS + s) * DD + tid];
    if (tid == 0) sMask = (int)mask[((size_t)b * SS + s) * LL];
    if (tid < 16) sNorm[tid] = 0.f;
    __syncthreads();

    // ---- gate[row] = sigmoid( e . (h[row] + keys[row]) ) ----
    float g = 0.f;
#pragma unroll
    for (int i = 0; i < 16; ++i) {
      const int c = gl + 16 * i;
      g += (sH[grow * HSTR + c] + kreg[i]) * sE[c];
    }
#pragma unroll
    for (int off = 8; off; off >>= 1) g += __shfl_xor(g, off, 16);
    if (gl == 0) sGate[grow] = 1.f / (1.f + expf(-g));
    __syncthreads();

    // ---- h @ U via bf16 WMMA (f32 accumulate) ----
    floatx8 acc0, acc1;
    wmma_row_tile(sH, sUt, lane, wave, acc0, acc1);

    // ---- h_tilda = relu(hU + keysV + eW); upd = h + gate*h_tilda ----
    float upd0[8], upd1[8], ho0[8], ho1[8], psum[8];
#pragma unroll
    for (int j = 0; j < 8; ++j) {
      const int m  = j + 8 * half;
      const int n0 = wave * 32 + nl, n1 = n0 + 16;
      const float gt = sGate[m];
      const float t0 = fmaxf(acc0[j] + kv0[j] + sEW[n0], 0.f);
      const float t1 = fmaxf(acc1[j] + kv1[j] + sEW[n1], 0.f);
      ho0[j] = sH[m * HSTR + n0];
      ho1[j] = sH[m * HSTR + n1];
      upd0[j] = ho0[j] + gt * t0;
      upd1[j] = ho1[j] + gt * t1;
      psum[j] = upd0[j] * upd0[j] + upd1[j] * upd1[j];
    }
    // per-row sum of squares: 16-lane shuffle reduce, then LDS f32 atomics
#pragma unroll
    for (int j = 0; j < 8; ++j) {
#pragma unroll
      for (int off = 8; off; off >>= 1) psum[j] += __shfl_xor(psum[j], off, 16);
    }
    if (nl == 0) {
#pragma unroll
      for (int j = 0; j < 8; ++j) atomicAdd(&sNorm[j + 8 * half], psum[j]);
    }
    __syncthreads();

    // ---- normalize, apply sentence mask, write back h ----
    const int msk = sMask;
#pragma unroll
    for (int j = 0; j < 8; ++j) {
      const int m  = j + 8 * half;
      const int n0 = wave * 32 + nl, n1 = n0 + 16;
      const float inv = rsqrtf(fmaxf(sNorm[m], 1e-12f));
      sH[m * HSTR + n0] = msk ? upd0[j] * inv : ho0[j];
      sH[m * HSTR + n1] = msk ? upd1[j] * inv : ho1[j];
    }
    __syncthreads();
  }

  // ---- write h_final ----
  for (int idx = tid; idx < 16 * DD; idx += 256) {
    const int r = idx >> 8, c = idx & 255;
    out[(size_t)(row0 + r) * DD + c] = sH[r * HSTR + c];
  }
}

// ---------------------------------------------------------------------------
// Launch
// ---------------------------------------------------------------------------
extern "C" void kernel_launch(void* const* d_in, const int* in_sizes, int n_in,
                              void* d_out, int out_size, void* d_ws, size_t ws_size,
                              hipStream_t stream) {
  (void)in_sizes; (void)n_in; (void)out_size; (void)ws_size;
  const int*           prgrph = (const int*)d_in[0];
  const unsigned char* pmask  = (const unsigned char*)d_in[1]; // bool (1B)
  const float*         keys   = (const float*)d_in[2];
  const float*         emb    = (const float*)d_in[3];
  const float*         U      = (const float*)d_in[4];
  const float*         V      = (const float*)d_in[5];
  const float*         W      = (const float*)d_in[6];
  float*               out    = (float*)d_out;

  // Workspace layout (f32): enc (B*S*D) | eW (B*S*D) | keysV (B*K*D)
  float* enc   = (float*)d_ws;
  float* eW    = enc + (size_t)NSENT * DD;
  float* keysV = eW  + (size_t)NSENT * DD;

  // 1) embedding gather+sum
  embed_sum_kernel<<<dim3(NSENT), dim3(256), 0, stream>>>(prgrph, emb, enc);
  // 2) eW = enc @ W   (8192 x 256 x 256)
  gemm256_bf16_kernel<<<dim3(NSENT / 16), dim3(256), 0, stream>>>(enc, W, eW);
  // 3) keysV = keys @ V  (4096 x 256 x 256)
  gemm256_bf16_kernel<<<dim3(NROW / 16), dim3(256), 0, stream>>>(keys, V, keysV);
  // 4) 64-step recurrence, h resident in LDS, zero global sync
  entity_scan_kernel<<<dim3(NROW / 16), dim3(256), 0, stream>>>(
      keys, keysV, enc, eW, pmask, U, out);
}
